// GaussPrior_57234734186511
// MI455X (gfx1250) — matrix operand order repaired and checked by reading, not compiled, not run on hardware
//
#include <hip/hip_runtime.h>

typedef __bf16 bf16_t;
typedef __attribute__((ext_vector_type(16))) __bf16 v16bf;
typedef __attribute__((ext_vector_type(8)))  float  v8f;

#define B_SZ 512
#define T_SZ 20
#define NOBJ_ 36
#define D_SZ 2048
#define EMB_ 300
#define E_SZ 512
#define HM_ 1024
#define HO_ 256
#define KM0 832    /* pad(E+EMB=812) to /32 */
#define KO0 1344   /* pad(2E+EMB=1324) to /32 */

__device__ __forceinline__ bf16_t f2bf(float f) {
    union { float f; unsigned u; } v; v.f = f;
    unsigned r = v.u + 0x7FFFu + ((v.u >> 16) & 1u);
    union { unsigned short s; bf16_t b; } o; o.s = (unsigned short)(r >> 16);
    return o.b;
}
__device__ __forceinline__ float sigm(float x) { return 1.f / (1.f + __expf(-x)); }

union FragU { uint4 q[2]; v16bf v; };

// C(MxN,f32) = A(MxK bf16 rowmaj, lda) @ W(NxK bf16 rowmaj, ldw)^T
//              [+ add0[m*N+n]] [+ add1[(m%add1Mod)*N+n]] [+ bias[n]]
// Block tile 128(M) x 256(N), 8 waves of 64x64, K-step 32, double-buffered LDS.
// Requires: M % 128 == 0, N % 256 == 0, K % 32 == 0.
__global__ __launch_bounds__(256)
void wmma_gemm_bf16(const bf16_t* __restrict__ A, int lda,
                    const bf16_t* __restrict__ W, int ldw,
                    float* __restrict__ C, int N, int K,
                    const float* __restrict__ add0,
                    const float* __restrict__ add1, int add1Mod,
                    const float* __restrict__ bias)
{
    __shared__ __align__(16) bf16_t As[2][128][40];
    __shared__ __align__(16) bf16_t Bs[2][256][40];
    const int tid   = threadIdx.x;
    const int lane  = tid & 31;
    const int wave  = tid >> 5;
    const int waveM = wave >> 2;        // 0..1 -> 64 rows each
    const int waveN = wave & 3;         // 0..3 -> 64 cols each
    const int r     = lane & 15;
    const int hlf   = lane >> 4;
    const size_t blockM = (size_t)blockIdx.y * 128;
    const size_t blockN = (size_t)blockIdx.x * 256;

    v8f acc[4][4];
#pragma unroll
    for (int i = 0; i < 4; ++i)
#pragma unroll
        for (int j = 0; j < 4; ++j)
#pragma unroll
            for (int e = 0; e < 8; ++e) acc[i][j][e] = 0.f;

    auto load_tiles = [&](int k0, int buf) {
        // A tile: 128 rows x 32 bf16 = 512 x 16B chunks
#pragma unroll
        for (int it = 0; it < 2; ++it) {
            int chunk = tid + it * 256;
            int row   = chunk >> 2;
            int cc    = chunk & 3;
            *(uint4*)&As[buf][row][cc * 8] =
                *(const uint4*)(A + (blockM + row) * (size_t)lda + k0 + cc * 8);
        }
        // B tile (W rows): 256 rows x 32 bf16 = 1024 x 16B chunks
#pragma unroll
        for (int it = 0; it < 4; ++it) {
            int chunk = tid + it * 256;
            int row   = chunk >> 2;
            int cc    = chunk & 3;
            *(uint4*)&Bs[buf][row][cc * 8] =
                *(const uint4*)(W + (blockN + row) * (size_t)ldw + k0 + cc * 8);
        }
    };

    const int nk = K >> 5;
    load_tiles(0, 0);
    __syncthreads();

    for (int ki = 0; ki < nk; ++ki) {
        const int cur = ki & 1;
        if (ki + 1 < nk) load_tiles((ki + 1) << 5, cur ^ 1);

        // B 32x16 bf16, contiguous k order used here (same as A): lanes 0-15 K=0..15 ; lanes 16-31 K=16..31 (col = lane%16)
        FragU bfr[4];
#pragma unroll
        for (int j = 0; j < 4; ++j) {
            const uint4* p = (const uint4*)&Bs[cur][waveN * 64 + j * 16 + r][0];
            bfr[j].q[0] = p[hlf * 2];
            bfr[j].q[1] = p[hlf * 2 + 1];
        }
        // A 16x32 bf16, contiguous k order used here (same as B): lanes 0-15 K=0..15 ; lanes 16-31 K=16..31 (row = lane%16)
#pragma unroll
        for (int i = 0; i < 4; ++i) {
            FragU af;
            const uint4* p = (const uint4*)&As[cur][waveM * 64 + i * 16 + r][0];
            af.q[0] = p[hlf * 2];
            af.q[1] = p[hlf * 2 + 1];
#pragma unroll
            for (int j = 0; j < 4; ++j)
                acc[i][j] = __builtin_amdgcn_wmma_f32_16x16x32_bf16(
                    false, af.v, false, bfr[j].v, (short)0, acc[i][j], false, false);
        }
        __syncthreads();
    }

    // C/D 16x16 f32 layout: vgpr v, lanes 0-15 -> M=v, lanes 16-31 -> M=8+v; N=lane%16
#pragma unroll
    for (int i = 0; i < 4; ++i) {
#pragma unroll
        for (int j = 0; j < 4; ++j) {
            size_t n  = blockN + waveN * 64 + j * 16 + r;
            size_t m0 = blockM + waveM * 64 + i * 16 + hlf * 8;
            float bv = bias ? bias[n] : 0.f;
#pragma unroll
            for (int v = 0; v < 8; ++v) {
                size_t m = m0 + v;
                float val = acc[i][j][v] + bv;
                if (add0) val += add0[m * (size_t)N + n];
                if (add1) val += add1[(m % (size_t)add1Mod) * (size_t)N + n];
                C[m * (size_t)N + n] = val;
            }
        }
    }
}

// per-(b,obj) mask: any |obj_enc| > 0 over D
__global__ void objmask_kernel(const float* __restrict__ obj_enc, float* __restrict__ om) {
    __shared__ float red[256];
    int bo = blockIdx.x;
    const float* p = obj_enc + (size_t)bo * D_SZ;
    float s = 0.f;
    for (int d = threadIdx.x; d < D_SZ; d += 256) s += fabsf(p[d]);
    red[threadIdx.x] = s; __syncthreads();
    for (int st = 128; st > 0; st >>= 1) {
        if (threadIdx.x < st) red[threadIdx.x] += red[threadIdx.x + st];
        __syncthreads();
    }
    if (threadIdx.x == 0) om[bo] = (red[0] > 0.f) ? 1.f : 0.f;
}

// meanimfeats (B x D) in bf16
__global__ void mif_kernel(const float* __restrict__ obj_enc, const float* __restrict__ om,
                           bf16_t* __restrict__ mif) {
    int idx = blockIdx.x * 256 + threadIdx.x;   // b*D + d
    int b = idx >> 11, d = idx & 2047;
    const float* base = obj_enc + ((size_t)b * NOBJ_) * D_SZ + d;
    float cnt = 0.f, s = 0.f;
    for (int o = 0; o < NOBJ_; ++o) {
        float m = om[b * NOBJ_ + o];
        cnt += m;
        s += m * base[(size_t)o * D_SZ];
    }
    mif[idx] = f2bf(s / fmaxf(cnt, 1e-9f));
}

// Xm rows (t*B+b) x KM0: [0:512]=z_m_prev, [512:812]=emb[x], pad 0
__global__ void pack_xm(const float* __restrict__ qzm, const float* __restrict__ emd,
                        const int* __restrict__ x, bf16_t* __restrict__ Xm) {
    int idx = blockIdx.x * 256 + threadIdx.x;   // < 10240*KM0
    int rrow = idx / KM0, c = idx % KM0;
    int t = rrow / B_SZ, b = rrow % B_SZ;
    float v = 0.f;
    if (c < E_SZ) {
        if (t > 0) v = qzm[(((size_t)b * T_SZ) + (t - 1)) * E_SZ + c];
    } else if (c < E_SZ + EMB_) {
        int tok = x[b * T_SZ + t];
        v = emd[(size_t)tok * EMB_ + (c - E_SZ)];
    }
    Xm[idx] = f2bf(v);
}

// Xo rows (t*B+b) x KO0: [0:512]=q_z_mask[t], [512:1024]=z_o_prev, [1024:1324]=emb, pad 0
__global__ void pack_xo(const float* __restrict__ qzm, const float* __restrict__ qzo,
                        const float* __restrict__ emd, const int* __restrict__ x,
                        bf16_t* __restrict__ Xo) {
    int idx = blockIdx.x * 256 + threadIdx.x;   // < 10240*KO0
    int rrow = idx / KO0, c = idx % KO0;
    int t = rrow / B_SZ, b = rrow % B_SZ;
    float v = 0.f;
    if (c < E_SZ) {
        v = qzm[(((size_t)b * T_SZ) + t) * E_SZ + c];
    } else if (c < 2 * E_SZ) {
        if (t > 0) v = qzo[(((size_t)b * T_SZ) + (t - 1)) * E_SZ + (c - E_SZ)];
    } else if (c < 2 * E_SZ + EMB_) {
        int tok = x[b * T_SZ + t];
        v = emd[(size_t)tok * EMB_ + (c - 2 * E_SZ)];
    }
    Xo[idx] = f2bf(v);
}

// f32 weight (N x srcLD), take cols [colOff, colOff+Kin) -> bf16 (N x Kpad) zero-padded
__global__ void cvt_w(const float* __restrict__ src, int srcLD, int colOff, int Kin,
                      bf16_t* __restrict__ dst, int Kpad, int total) {
    int idx = blockIdx.x * 256 + threadIdx.x;
    if (idx >= total) return;
    int n = idx / Kpad, c = idx % Kpad;
    float v = (c < Kin) ? src[(size_t)n * srcLD + colOff + c] : 0.f;
    dst[idx] = f2bf(v);
}

// gates (B x 4H): [i f g o] -> c, h(bf16)
__global__ void lstm_cell(const float* __restrict__ gates, float* __restrict__ c,
                          bf16_t* __restrict__ h_bf, int H) {
    int idx = blockIdx.x * 256 + threadIdx.x;   // < B*H
    int b = idx / H, j = idx % H;
    const float* g = gates + (size_t)b * 4 * H;
    float ig = sigm(g[j]);
    float fg = sigm(g[H + j]);
    float gg = tanhf(g[2 * H + j]);
    float og = sigm(g[3 * H + j]);
    float cn = fg * c[idx] + ig * gg;
    c[idx] = cn;
    h_bf[idx] = f2bf(og * tanhf(cn));
}

// fused KL(t): sum_E for mask + obj branches, * seq_mask, accumulate into out[b]
__global__ void kl_kernel(const float* __restrict__ out_m, const float* __restrict__ out_o,
                          const float* __restrict__ qmm, const float* __restrict__ qlm,
                          const float* __restrict__ qmo, const float* __restrict__ qlo,
                          const int* __restrict__ x, int t, float* __restrict__ out) {
    __shared__ float red[256];
    int b = blockIdx.x;
    int tok = x[b * T_SZ + t];
    float smask = ((tok != 0) && (tok != 2)) ? 1.f : 0.f;
    size_t qb = ((size_t)b * T_SZ + t) * E_SZ;
    float s = 0.f;
    for (int e = threadIdx.x; e < E_SZ; e += 256) {
        float pm = out_m[(size_t)b * 1024 + e];
        float pl = out_m[(size_t)b * 1024 + E_SZ + e];
        float qm = qmm[qb + e], ql = qlm[qb + e];
        float dm = qm - pm;
        s += 0.5f * (pl - ql) + (__expf(ql) + dm * dm) / (2.f * __expf(pl)) - 0.5f;
        pm = out_o[(size_t)b * 1024 + e];
        pl = out_o[(size_t)b * 1024 + E_SZ + e];
        qm = qmo[qb + e]; ql = qlo[qb + e];
        dm = qm - pm;
        s += 0.5f * (pl - ql) + (__expf(ql) + dm * dm) / (2.f * __expf(pl)) - 0.5f;
    }
    red[threadIdx.x] = s; __syncthreads();
    for (int st = 128; st > 0; st >>= 1) {
        if (threadIdx.x < st) red[threadIdx.x] += red[threadIdx.x + st];
        __syncthreads();
    }
    if (threadIdx.x == 0) out[b] += smask * red[0];
}

extern "C" void kernel_launch(void* const* d_in, const int* in_sizes, int n_in,
                              void* d_out, int out_size, void* d_ws, size_t ws_size,
                              hipStream_t stream) {
    const float* obj_enc = (const float*)d_in[0];
    const int*   x       = (const int*)d_in[1];
    const float* qmm     = (const float*)d_in[2];
    const float* qlm     = (const float*)d_in[3];
    const float* qzm     = (const float*)d_in[4];
    const float* qmo     = (const float*)d_in[5];
    const float* qlo     = (const float*)d_in[6];
    const float* qzo     = (const float*)d_in[7];
    const float* emd     = (const float*)d_in[8];
    const float* Wihm    = (const float*)d_in[9];
    const float* Whhm    = (const float*)d_in[10];
    const float* bm      = (const float*)d_in[11];
    const float* Wiho    = (const float*)d_in[12];
    const float* Whho    = (const float*)d_in[13];
    const float* bo      = (const float*)d_in[14];
    const float* fcmW    = (const float*)d_in[15];
    const float* fcmb    = (const float*)d_in[16];
    const float* fcoW    = (const float*)d_in[17];
    const float* fcob    = (const float*)d_in[18];

    char* ws = (char*)d_ws;
    size_t off = 0;
    auto alloc = [&](size_t bytes) -> void* {
        void* p = ws + off;
        off = (off + bytes + 255) & ~(size_t)255;
        return p;
    };
    const size_t BT = (size_t)B_SZ * T_SZ;      // 10240

    bf16_t* mif_bf = (bf16_t*)alloc((size_t)B_SZ * D_SZ * 2);
    float*  om     = (float*) alloc((size_t)B_SZ * NOBJ_ * 4);
    bf16_t* Xm     = (bf16_t*)alloc(BT * KM0 * 2);
    bf16_t* Xo     = (bf16_t*)alloc(BT * KO0 * 2);
    bf16_t* Wm0    = (bf16_t*)alloc((size_t)4 * HM_ * KM0 * 2);
    bf16_t* Wmifm  = (bf16_t*)alloc((size_t)4 * HM_ * D_SZ * 2);
    bf16_t* Wo0    = (bf16_t*)alloc((size_t)4 * HO_ * KO0 * 2);
    bf16_t* Wmifo  = (bf16_t*)alloc((size_t)4 * HO_ * D_SZ * 2);
    bf16_t* Whhm_b = (bf16_t*)alloc((size_t)4 * HM_ * HM_ * 2);
    bf16_t* Whho_b = (bf16_t*)alloc((size_t)4 * HO_ * HO_ * 2);
    bf16_t* fcm_b  = (bf16_t*)alloc((size_t)2 * E_SZ * HM_ * 2);
    bf16_t* fco_b  = (bf16_t*)alloc((size_t)2 * E_SZ * HO_ * 2);
    float*  Gbm    = (float*) alloc((size_t)B_SZ * 4 * HM_ * 4);
    float*  Gbo    = (float*) alloc((size_t)B_SZ * 4 * HO_ * 4);
    float*  Gm     = (float*) alloc(BT * 4 * HM_ * 4);
    float*  Go     = (float*) alloc(BT * 4 * HO_ * 4);
    float*  gts_m  = (float*) alloc((size_t)B_SZ * 4 * HM_ * 4);
    float*  gts_o  = (float*) alloc((size_t)B_SZ * 4 * HO_ * 4);
    bf16_t* h_m    = (bf16_t*)alloc((size_t)B_SZ * HM_ * 2);
    float*  c_m    = (float*) alloc((size_t)B_SZ * HM_ * 4);
    bf16_t* h_o    = (bf16_t*)alloc((size_t)B_SZ * HO_ * 2);
    float*  c_o    = (float*) alloc((size_t)B_SZ * HO_ * 4);
    float*  o_m    = (float*) alloc((size_t)B_SZ * 2 * E_SZ * 4);
    float*  o_o    = (float*) alloc((size_t)B_SZ * 2 * E_SZ * 4);
    float*  outp   = (float*)d_out;

    // zero-init state + output accumulator (graph-capturable memset nodes)
    hipMemsetAsync(h_m, 0, (size_t)B_SZ * HM_ * 2, stream);
    hipMemsetAsync(c_m, 0, (size_t)B_SZ * HM_ * 4, stream);
    hipMemsetAsync(h_o, 0, (size_t)B_SZ * HO_ * 2, stream);
    hipMemsetAsync(c_o, 0, (size_t)B_SZ * HO_ * 4, stream);
    hipMemsetAsync(outp, 0, (size_t)B_SZ * 4, stream);

    // ---- prologue: mif, packed activations, bf16 weights
    objmask_kernel<<<B_SZ * NOBJ_, 256, 0, stream>>>(obj_enc, om);
    mif_kernel<<<(B_SZ * D_SZ) / 256, 256, 0, stream>>>(obj_enc, om, mif_bf);
    pack_xm<<<(int)(BT * KM0 / 256), 256, 0, stream>>>(qzm, emd, x, Xm);
    pack_xo<<<(int)(BT * KO0 / 256), 256, 0, stream>>>(qzm, qzo, emd, x, Xo);

    auto cvt = [&](const float* src, int ld, int coff, int kin, bf16_t* dst, int kpad, int nrows) {
        int total = nrows * kpad;
        cvt_w<<<(total + 255) / 256, 256, 0, stream>>>(src, ld, coff, kin, dst, kpad, total);
    };
    cvt(Wihm, 2860, 0,    812,  Wm0,    KM0,  4 * HM_);
    cvt(Wihm, 2860, 812,  2048, Wmifm,  D_SZ, 4 * HM_);
    cvt(Wiho, 3372, 0,    1324, Wo0,    KO0,  4 * HO_);
    cvt(Wiho, 3372, 1324, 2048, Wmifo,  D_SZ, 4 * HO_);
    cvt(Whhm, HM_,  0,    HM_,  Whhm_b, HM_,  4 * HM_);
    cvt(Whho, HO_,  0,    HO_,  Whho_b, HO_,  4 * HO_);
    cvt(fcmW, HM_,  0,    HM_,  fcm_b,  HM_,  2 * E_SZ);
    cvt(fcoW, HO_,  0,    HO_,  fco_b,  HO_,  2 * E_SZ);

    // ---- time-invariant base projections: Gbase = mif @ W_mif^T + b
    wmma_gemm_bf16<<<dim3(4 * HM_ / 256, B_SZ / 128), 256, 0, stream>>>(
        mif_bf, D_SZ, Wmifm, D_SZ, Gbm, 4 * HM_, D_SZ, nullptr, nullptr, 1, bm);
    wmma_gemm_bf16<<<dim3(4 * HO_ / 256, B_SZ / 128), 256, 0, stream>>>(
        mif_bf, D_SZ, Wmifo, D_SZ, Gbo, 4 * HO_, D_SZ, nullptr, nullptr, 1, bo);

    // ---- time-varying input projections for ALL steps (rows = t*B+b), += Gbase[b]
    wmma_gemm_bf16<<<dim3(4 * HM_ / 256, (int)(BT / 128)), 256, 0, stream>>>(
        Xm, KM0, Wm0, KM0, Gm, 4 * HM_, KM0, nullptr, Gbm, B_SZ, nullptr);
    wmma_gemm_bf16<<<dim3(4 * HO_ / 256, (int)(BT / 128)), 256, 0, stream>>>(
        Xo, KO0, Wo0, KO0, Go, 4 * HO_, KO0, nullptr, Gbo, B_SZ, nullptr);

    // ---- sequential recurrence
    for (int t = 0; t < T_SZ; ++t) {
        // mask LSTM: gates = G_m[t] + h_m @ W_hh_m^T ; cell
        wmma_gemm_bf16<<<dim3(4 * HM_ / 256, B_SZ / 128), 256, 0, stream>>>(
            h_m, HM_, Whhm_b, HM_, gts_m, 4 * HM_, HM_,
            Gm + (size_t)t * B_SZ * 4 * HM_, nullptr, 1, nullptr);
        lstm_cell<<<(B_SZ * HM_) / 256, 256, 0, stream>>>(gts_m, c_m, h_m, HM_);

        // obj LSTM
        wmma_gemm_bf16<<<dim3(4 * HO_ / 256, B_SZ / 128), 256, 0, stream>>>(
            h_o, HO_, Whho_b, HO_, gts_o, 4 * HO_, HO_,
            Go + (size_t)t * B_SZ * 4 * HO_, nullptr, 1, nullptr);
        lstm_cell<<<(B_SZ * HO_) / 256, 256, 0, stream>>>(gts_o, c_o, h_o, HO_);

        // fc heads
        wmma_gemm_bf16<<<dim3(2 * E_SZ / 256, B_SZ / 128), 256, 0, stream>>>(
            h_m, HM_, fcm_b, HM_, o_m, 2 * E_SZ, HM_, nullptr, nullptr, 1, fcmb);
        wmma_gemm_bf16<<<dim3(2 * E_SZ / 256, B_SZ / 128), 256, 0, stream>>>(
            h_o, HO_, fco_b, HO_, o_o, 2 * E_SZ, HO_, nullptr, nullptr, 1, fcob);

        // fused KL accumulation into d_out
        kl_kernel<<<B_SZ, 256, 0, stream>>>(o_m, o_o, qmm, qlm, qmo, qlo, x, t, outp);
    }
    (void)in_sizes; (void)n_in; (void)out_size; (void)ws_size;
}
